// RNN_40003325395438
// MI455X (gfx1250) — matrix operand order repaired and checked by reading, not compiled, not run on hardware
//
#include <hip/hip_runtime.h>

// Problem dims (fixed by the reference)
#define BB 64     // batch
#define TT 2048   // timesteps
#define II 64     // input size
#define HH 128    // hidden size
#define HP 136    // padded LDS row stride in f16 (272B = 17*16B: aligned, conflict-free)

typedef __attribute__((ext_vector_type(16))) _Float16 v16h;
typedef __attribute__((ext_vector_type(8)))  _Float16 v8h;
typedef __attribute__((ext_vector_type(8)))  float    v8f;

// tanh(x) = 1 - 2/(1 + e^{2x}); exact saturation at +-inf, ~2 ulp core error.
// 2 trans ops (v_exp_f32, v_rcp_f32) + 3 VALU vs. the OCML polynomial.
__device__ __forceinline__ float fast_tanh(float x) {
  float e = __expf(2.0f * x);
  float r = __builtin_amdgcn_rcpf(1.0f + e);
  return __builtin_fmaf(-2.0f, r, 1.0f);
}

// ---------------------------------------------------------------------------
// Kernel 1: xw[b,t,:] = x[b,t,:] @ W^T + (b_ih + b_hh), written into d_out.
// GEMM  M = B*T = 131072, N = H = 128, K = I = 64. Memory-bound (~100 MB).
// Block = 256 threads = 8 waves; wave w owns N-tile w; block owns 16 M-rows.
// ---------------------------------------------------------------------------
__global__ __launch_bounds__(256) void rnn_xw_kernel(
    const float* __restrict__ x, const float* __restrict__ W,
    const float* __restrict__ b_ih, const float* __restrict__ b_hh,
    float* __restrict__ out)
{
  const int lane = threadIdx.x & 31;
  const int wave = threadIdx.x >> 5;      // N-tile index 0..7
  const int m    = lane & 15;
  const int half = lane >> 4;
  const int rowBase = blockIdx.x * 16;    // M-tile over B*T rows
  const int nCol = wave * 16 + m;
  const float bsum = b_ih[nCol] + b_hh[nCol];

  v8f c = {};
#pragma unroll
  for (int ks = 0; ks < 2; ++ks) {        // K = 64 -> 2 steps of 32
    // A fragment (16x32 f16): lane = row lane%16; K chunks at
    // ks*32 + half*8 + {0..7} and {16..23}  (ISA 16-bit A layout)
    const float* ap = x + (size_t)(rowBase + m) * II + ks * 32 + half * 8;
    v16h a;
#pragma unroll
    for (int j = 0; j < 8; ++j) {
      a[j]     = (_Float16)ap[j];
      a[j + 8] = (_Float16)ap[j + 16];
    }
    // B fragment (32x16 f16): B[k][n] = W[n][k]; lane = col lane%16,
    // K chunks at ks*32 + half*8 + {0..7} and {16..23} (same as A).
    const float* bp = W + (size_t)nCol * II + ks * 32 + half * 8;
    v16h b;
#pragma unroll
    for (int j = 0; j < 8; ++j) {
      b[j]     = (_Float16)bp[j];
      b[j + 8] = (_Float16)bp[j + 16];
    }

    c = __builtin_amdgcn_wmma_f32_16x16x32_f16(false, a, false, b,
                                               (short)0, c, false, false);
  }
  // D layout: lane L -> col L%16, rows (L/16)*8 + v
#pragma unroll
  for (int v = 0; v < 8; ++v) {
    const size_t row = (size_t)rowBase + half * 8 + v;
    out[row * HH + nCol] = c[v] + bsum;
  }
}

// ---------------------------------------------------------------------------
// Kernel 2: sequential scan  h = tanh(xw_t + h @ U^T).
// 4 blocks (16 batch rows) x 8 waves (16 H-cols) = all 32 WMMA tiles.
// Per-step critical path: 8x ds_load_b128 -> two independent 2-WMMA chains
// -> 8 adds -> fast tanh -> f16 LDS publish -> 1 barrier. Global stores for
// h_t retire after the barrier, under the next step's compute. U's
// B-fragments are loop-invariant registers; xt(t+1) loads issue before the
// WMMAs so they fly under the matrix ops.
// ---------------------------------------------------------------------------
__global__ __launch_bounds__(256) void rnn_scan_kernel(
    const float* __restrict__ h0, const float* __restrict__ U,
    float* __restrict__ out /* [B,T,H]: holds xw, overwritten with h */,
    float* __restrict__ hlast /* [B,H] */)
{
  __shared__ _Float16 hsh[2][16 * HP];    // double-buffered h tile (8.5 KB)

  const int lane = threadIdx.x & 31;
  const int wave = threadIdx.x >> 5;
  const int m    = lane & 15;
  const int half = lane >> 4;
  const int bBase = blockIdx.x * 16;
  const int nCol  = wave * 16 + m;

  // Preload U as B fragments (loop-invariant): B[k][n] = U[n][k]
  v16h ub[4];
#pragma unroll
  for (int ks = 0; ks < 4; ++ks) {        // K = 128 -> 4 chunks of 32
    const float* up = U + (size_t)nCol * HH + ks * 32 + half * 8;
#pragma unroll
    for (int j = 0; j < 8; ++j) {
      ub[ks][j]     = (_Float16)up[j];
      ub[ks][j + 8] = (_Float16)up[j + 16];
    }
  }

  // Initialize h buffer 0 from h0 (f32 -> f16)
  for (int idx = threadIdx.x; idx < 16 * HH; idx += 256) {
    const int mm = idx >> 7;
    const int hh = idx & (HH - 1);
    hsh[0][mm * HP + hh] = (_Float16)h0[(bBase + mm) * HH + hh];
  }
  __syncthreads();

  const int rowTop = bBase + half * 8;    // first batch row this lane covers

  // Prefetch xt for t = 0
  v8f xt;
#pragma unroll
  for (int v = 0; v < 8; ++v)
    xt[v] = out[((size_t)(rowTop + v) * TT) * HH + nCol];

  int cur = 0;
  v8f d = {};
  for (int t = 0; t < TT; ++t) {
    // A fragments of h from LDS: two aligned 16B chunks per k-step
    // (row stride 272B -> ds_load_b128, bank-conflict-free).
    v16h a[4];
#pragma unroll
    for (int ks = 0; ks < 4; ++ks) {
      const v8h* p = (const v8h*)&hsh[cur][m * HP + ks * 32 + half * 8];
      v8h lo = p[0];                      // K offsets +0..7
      v8h hi = p[2];                      // K offsets +16..23 (16 halves on)
      a[ks] = __builtin_shufflevector(lo, hi, 0, 1, 2, 3, 4, 5, 6, 7,
                                      8, 9, 10, 11, 12, 13, 14, 15);
    }

    // Issue next timestep's xt loads before the WMMAs (fly under compute)
    v8f xtn;
    if (t + 1 < TT) {
#pragma unroll
      for (int v = 0; v < 8; ++v)
        xtn[v] = out[((size_t)(rowTop + v) * TT + (t + 1)) * HH + nCol];
    }

    // Two independent accumulation chains -> half the WMMA RAW depth
    v8f zero = {};
    v8f c0 = __builtin_amdgcn_wmma_f32_16x16x32_f16(false, a[0], false, ub[0],
                                                    (short)0, xt, false, false);
    v8f c1 = __builtin_amdgcn_wmma_f32_16x16x32_f16(false, a[1], false, ub[1],
                                                    (short)0, zero, false, false);
    c0 = __builtin_amdgcn_wmma_f32_16x16x32_f16(false, a[2], false, ub[2],
                                                (short)0, c0, false, false);
    c1 = __builtin_amdgcn_wmma_f32_16x16x32_f16(false, a[3], false, ub[3],
                                                (short)0, c1, false, false);

#pragma unroll
    for (int v = 0; v < 8; ++v) d[v] = fast_tanh(c0[v] + c1[v]);

    // Publish h_t (f16) to the other LDS buffer FIRST -- this is what the
    // other waves are waiting on at the barrier.
#pragma unroll
    for (int v = 0; v < 8; ++v)
      hsh[cur ^ 1][(half * 8 + v) * HP + nCol] = (_Float16)d[v];
    __syncthreads();

    // Emit output h_t after the barrier; stores retire under the next step.
#pragma unroll
    for (int v = 0; v < 8; ++v)
      out[((size_t)(rowTop + v) * TT + t) * HH + nCol] = d[v];

    xt = xtn;
    cur ^= 1;
  }

  // h_last = h_{T-1}
#pragma unroll
  for (int v = 0; v < 8; ++v)
    hlast[(size_t)(rowTop + v) * HH + nCol] = d[v];
}

// ---------------------------------------------------------------------------
extern "C" void kernel_launch(void* const* d_in, const int* in_sizes, int n_in,
                              void* d_out, int out_size, void* d_ws, size_t ws_size,
                              hipStream_t stream) {
  const float* x    = (const float*)d_in[0];
  const float* h0   = (const float*)d_in[1];
  const float* W    = (const float*)d_in[2];
  const float* U    = (const float*)d_in[3];
  const float* b_ih = (const float*)d_in[4];
  const float* b_hh = (const float*)d_in[5];

  float* out   = (float*)d_out;                  // [B,T,H]
  float* hlast = out + (size_t)BB * TT * HH;     // [B,H]

  // Phase 1: memory-bound WMMA GEMM fills d_out with xw (fully rewritten
  // every call -> deterministic across graph replays).
  rnn_xw_kernel<<<dim3((BB * TT) / 16), dim3(256), 0, stream>>>(
      x, W, b_ih, b_hh, out);

  // Phase 2: latency-bound recurrence; 4 workgroups cover all B x H tiles.
  rnn_scan_kernel<<<dim3(BB / 16), dim3(256), 0, stream>>>(h0, U, out, hlast);

  (void)in_sizes; (void)n_in; (void)out_size; (void)d_ws; (void)ws_size;
}
